// NeurJudge_plus_30056181138114
// MI455X (gfx1250) — compile-verified, matrix operand order checked by reading, not run on hardware
//
#include <hip/hip_runtime.h>
#include <hip/hip_bf16.h>
#include <math.h>
#include <stdint.h>

typedef __attribute__((ext_vector_type(16))) _Float16 v16h;
typedef __attribute__((ext_vector_type(8)))  _Float16 v8h;
typedef __attribute__((ext_vector_type(8)))  float    v8f;

#define CV   50000
#define CD   200
#define CHID 150
#define CNC  119
#define CNA  103
#define CNT  11
#define CB   32
#define CT   512
#define CLC  50
#define CLV  50
#define CEC  600
#define CEA  500

static inline int ceil16i(int x){ return (x + 15) & ~15; }
static inline int ceil32i(int x){ return (x + 31) & ~31; }
static inline int ceil64i(int x){ return (x + 63) & ~63; }

// ---------------------------------------------------------------------------
// WMMA fragment load: 16 halves per lane.
// A(16x32 f16) layout: lane L: row = L&15; halves[0..7] = K = 8*(L>>4)+0..7,
// halves[8..15] = K = 16 + 8*(L>>4) + 0..7.  B loaded symmetrically (N,K).
// ---------------------------------------------------------------------------
__device__ __forceinline__ v16h ldfrag16(const _Float16* p) {
  v8h lo = *(const v8h*)(p);
  v8h hi = *(const v8h*)(p + 16);
  return __builtin_shufflevector(lo, hi, 0,1,2,3,4,5,6,7,8,9,10,11,12,13,14,15);
}

// ---------------------------------------------------------------------------
// Batched WMMA GEMM:  C[M,N] = A[M,Kp] * B[N,Kp]^T (+ bias[n])
//  * A,B f16; A rows padded to 16 and lda == Kp (contiguous 16-row panels),
//    B rows padded to 64, Kp multiple of 32, zero-filled padding.
//  * Block = 4 waves, all sharing one 16-row A panel, staged to LDS with
//    CDNA5 async DMA (global_load_async_to_lds_b128 / s_wait_asynccnt).
//  * Each wave computes a 16x64 strip: one A fragment reused across 4 B
//    fragments -> 4 v_wmma per A-fragment load.
// ---------------------------------------------------------------------------
__global__ void k_gemm(const _Float16* __restrict__ A, const _Float16* __restrict__ Bm,
                       const float* __restrict__ bias, float* __restrict__ C,
                       int M, int N, int Kp, int lda, int ldb, int ldc,
                       long sA, long sB, long sC)
{
  extern __shared__ char smemg[];
  _Float16* Atile = (_Float16*)smemg;                 // 16 * Kp halves
  int lane = threadIdx.x;
  int m0 = blockIdx.y * 16;
  long bz = blockIdx.z;
  const _Float16* Ab = A + bz * sA;
  const _Float16* Bb = Bm + bz * sB;
  float* Cb = C + bz * sC;

  // --- async-DMA the shared A panel (contiguous: lda == Kp) into LDS ---
  {
    int tid = threadIdx.x + threadIdx.y * 32;
    int nthr = blockDim.x * blockDim.y;
    const _Float16* Asrc = Ab + (long)m0 * lda;
    unsigned ldsbase = (unsigned)(uintptr_t)Atile;
    unsigned nch = 2u * (unsigned)Kp;                 // 16-byte chunks
    for (unsigned c = tid; c < nch; c += (unsigned)nthr) {
      unsigned loff = ldsbase + c * 16u;
      unsigned long long ga = (unsigned long long)(uintptr_t)Asrc
                            + (unsigned long long)c * 16ull;
      asm volatile("global_load_async_to_lds_b128 %0, %1, off"
                   :: "v"(loff), "v"(ga) : "memory");
    }
    asm volatile("s_wait_asynccnt 0" ::: "memory");
  }
  __syncthreads();

  int Nc = (N + 63) & ~63;
  int n0 = (blockIdx.x * blockDim.y + threadIdx.y) * 64;
  if (n0 >= Nc) return;                               // uniform per wave

  int half = lane >> 4;
  const _Float16* aL = Atile + (long)(lane & 15) * Kp + half * 8;
  const _Float16* bp0 = Bb + (long)(n0 +  0 + (lane & 15)) * ldb + half * 8;
  const _Float16* bp1 = Bb + (long)(n0 + 16 + (lane & 15)) * ldb + half * 8;
  const _Float16* bp2 = Bb + (long)(n0 + 32 + (lane & 15)) * ldb + half * 8;
  const _Float16* bp3 = Bb + (long)(n0 + 48 + (lane & 15)) * ldb + half * 8;

  v8f acc[4];
#pragma unroll
  for (int j = 0; j < 4; ++j) acc[j] = (v8f){0.f,0.f,0.f,0.f,0.f,0.f,0.f,0.f};

  for (int k0 = 0; k0 < Kp; k0 += 32) {
    v16h af  = ldfrag16(aL + k0);                     // LDS (ds_load_b128)
    v16h bf0 = ldfrag16(bp0 + k0);
    v16h bf1 = ldfrag16(bp1 + k0);
    v16h bf2 = ldfrag16(bp2 + k0);
    v16h bf3 = ldfrag16(bp3 + k0);
    acc[0] = __builtin_amdgcn_wmma_f32_16x16x32_f16(false, af, false, bf0, (short)0, acc[0], false, false);
    acc[1] = __builtin_amdgcn_wmma_f32_16x16x32_f16(false, af, false, bf1, (short)0, acc[1], false, false);
    acc[2] = __builtin_amdgcn_wmma_f32_16x16x32_f16(false, af, false, bf2, (short)0, acc[2], false, false);
    acc[3] = __builtin_amdgcn_wmma_f32_16x16x32_f16(false, af, false, bf3, (short)0, acc[3], false, false);
  }

#pragma unroll
  for (int j = 0; j < 4; ++j) {
    int n = n0 + 16 * j + (lane & 15);
    if (n < N) {
      float bv = bias ? bias[n] : 0.0f;
#pragma unroll
      for (int r = 0; r < 8; ++r) {
        int m = m0 + r + half * 8;
        if (m < M) Cb[(long)m * ldc + n] = acc[j][r] + bv;
      }
    }
  }
}

// ---------------------------------------------------------------------------
// GRU scan (one block per direction, 32 waves).  h kept in LDS as f16;
// per step gh = h @ Whh^T via WMMA from LDS, gates in VALU.
// xs is precomputed x@Wih^T+bih, f32 (Bseq*T, 3Hd).
// ys: (Bseq, T, 2*Hd); dir 0 -> cols [0,Hd), dir 1 -> cols [Hd,2Hd).
// ---------------------------------------------------------------------------
__global__ void k_gru_scan(const float* __restrict__ xs_f, const float* __restrict__ xs_b,
                           const _Float16* __restrict__ Whh_f, const _Float16* __restrict__ Whh_b,
                           const float* __restrict__ bhh2,   // (2, 3Hd)
                           float* __restrict__ ys,
                           int Bseq, int T, int Hd, int Kp, int Np)
{
  extern __shared__ char smem[];
  int Bpad = (Bseq + 15) & ~15;
  _Float16* h16 = (_Float16*)smem;                        // Bpad * Kp
  float* gpre = (float*)(smem + (size_t)Bpad * Kp * 2);   // Bpad * Np
  int dir = blockIdx.x;
  const float* xs = dir ? xs_b : xs_f;
  const _Float16* Whh = dir ? Whh_b : Whh_f;
  const float* bh = bhh2 + (long)dir * 3 * Hd;
  int H3 = 3 * Hd;
  int lane = threadIdx.x;
  int wave = threadIdx.y;
  int nwaves = blockDim.y;
  int tid = lane + wave * 32;
  int nthreads = 32 * nwaves;
  for (int i = tid; i < Bpad * Kp; i += nthreads) h16[i] = (_Float16)0.f;
  __syncthreads();
  int mtiles = Bpad / 16, ntiles = Np / 16;
  int tot = mtiles * ntiles;
  int half = lane >> 4;
  for (int step = 0; step < T; ++step) {
    int t = dir ? (T - 1 - step) : step;
    for (int tile = wave; tile < tot; tile += nwaves) {
      int tm = tile / ntiles, tn = tile - tm * ntiles;
      int m0 = tm * 16, n0 = tn * 16;
      const _Float16* ap = h16 + (long)(m0 + (lane & 15)) * Kp + half * 8;
      const _Float16* bp = Whh + (long)(n0 + (lane & 15)) * Kp + half * 8;
      v8f acc = {0.f,0.f,0.f,0.f,0.f,0.f,0.f,0.f};
      for (int k0 = 0; k0 < Kp; k0 += 32) {
        v16h af = ldfrag16(ap + k0);
        v16h bf = ldfrag16(bp + k0);
        acc = __builtin_amdgcn_wmma_f32_16x16x32_f16(false, af, false, bf,
                                                     (short)0, acc, false, false);
      }
      int n = n0 + (lane & 15);
#pragma unroll
      for (int r = 0; r < 8; ++r) {
        int m = m0 + r + half * 8;
        gpre[m * Np + n] = acc[r];
      }
    }
    __syncthreads();
    for (int i = tid; i < Bseq * Hd; i += nthreads) {
      int b = i / Hd, j = i - b * Hd;
      long row = (long)b * T + t;
      const float* xrow = xs + row * H3;
      float ghr = gpre[b * Np + j]          + bh[j];
      float ghz = gpre[b * Np + Hd + j]     + bh[Hd + j];
      float ghn = gpre[b * Np + 2 * Hd + j] + bh[2 * Hd + j];
      float r = 1.f / (1.f + expf(-(xrow[j] + ghr)));
      float z = 1.f / (1.f + expf(-(xrow[Hd + j] + ghz)));
      float nn = tanhf(xrow[2 * Hd + j] + r * ghn);
      float hprev = (float)h16[b * Kp + j];
      float hnew = (1.f - z) * nn + z * hprev;
      h16[b * Kp + j] = (_Float16)hnew;
      ys[row * (2 * Hd) + dir * Hd + j] = hnew;
    }
    __syncthreads();
  }
}

// --------------------------- staging / elementwise --------------------------
__global__ void k_pad16(const float* __restrict__ src, int R, int Cs,
                        _Float16* __restrict__ dst, int Rp, int Cp)
{
  long i = (long)blockIdx.x * blockDim.x + threadIdx.x;
  if (i >= (long)Rp * Cp) return;
  int r = (int)(i / Cp), c = (int)(i % Cp);
  float v = (r < R && c < Cs) ? src[(long)r * Cs + c] : 0.f;
  dst[i] = (_Float16)v;
}

__global__ void k_embed16(const float* __restrict__ E, const int* __restrict__ tok,
                          int Rtok, _Float16* __restrict__ dst, int Rp, int Cp)
{
  long i = (long)blockIdx.x * blockDim.x + threadIdx.x;
  if (i >= (long)Rp * Cp) return;
  int r = (int)(i / Cp), c = (int)(i % Cp);
  float v = 0.f;
  if (r < Rtok && c < CD) v = E[(long)tok[r] * CD + c];
  dst[i] = (_Float16)v;
}

__global__ void k_embed_pred16(const float* __restrict__ E, const int* __restrict__ vt,
                               const int* __restrict__ pred, int Lv,
                               _Float16* __restrict__ dst, int Rp, int Cp)
{
  long i = (long)blockIdx.x * blockDim.x + threadIdx.x;
  if (i >= (long)Rp * Cp) return;
  int r = (int)(i / Cp), c = (int)(i % Cp);
  float v = 0.f;
  int Rtok = CB * Lv;
  if (r < Rtok && c < CD) {
    int b = r / Lv, l = r - b * Lv;
    int tokv = vt[pred[b] * Lv + l];
    v = E[(long)tokv * CD + c];
  }
  dst[i] = (_Float16)v;
}

__global__ void k_pad_batch16(const float* __restrict__ src, int Bt, int R, int C,
                              _Float16* __restrict__ dst, int Rp, int Cp, int transpose)
{
  long i = (long)blockIdx.x * blockDim.x + threadIdx.x;
  long per = (long)Rp * Cp;
  if (i >= (long)Bt * per) return;
  int b = (int)(i / per);
  long rem = i - (long)b * per;
  int r = (int)(rem / Cp), c = (int)(rem % Cp);
  float v = 0.f;
  if (!transpose) { if (r < R && c < C) v = src[((long)b * R + r) * C + c]; }
  else            { if (r < C && c < R) v = src[((long)b * R + c) * C + r]; }
  dst[i] = (_Float16)v;
}

__global__ void k_cat16(const float* s0, int w0, int b0, const float* s1, int w1, int b1,
                        const float* s2, int w2, int b2, const float* s3, int w3, int b3,
                        int Mrows, int T, _Float16* __restrict__ dst, int Cp)
{
  long i = (long)blockIdx.x * blockDim.x + threadIdx.x;
  int Mp = (Mrows + 15) & ~15;
  if (i >= (long)Mp * Cp) return;
  int r = (int)(i / Cp), c = (int)(i % Cp);
  float v = 0.f;
  if (r < Mrows) {
    int rb = r / T;
    if (c < w0) v = b0 ? s0[(long)rb * w0 + c] : s0[(long)r * w0 + c];
    else {
      c -= w0;
      if (c < w1) v = b1 ? s1[(long)rb * w1 + c] : s1[(long)r * w1 + c];
      else {
        c -= w1;
        if (c < w2) v = b2 ? s2[(long)rb * w2 + c] : s2[(long)r * w2 + c];
        else {
          c -= w2;
          if (w3 > 0 && c < w3) v = b3 ? s3[(long)rb * w3 + c] : s3[(long)r * w3 + c];
        }
      }
    }
  }
  dst[i] = (_Float16)v;
}

__global__ void k_meanT(const float* __restrict__ in, int Bq, int T, int ld, int C,
                        float* __restrict__ out)
{
  int i = blockIdx.x * blockDim.x + threadIdx.x;
  if (i >= Bq * C) return;
  int b = i / C, c = i - b * C;
  float s = 0.f;
  for (int t = 0; t < T; ++t) s += in[((long)b * T + t) * ld + c];
  out[i] = s / (float)T;
}

// ------------------------------ graph decomp --------------------------------
__global__ void k_counts(const int* __restrict__ src, int E, float* __restrict__ counts)
{
  int e = blockIdx.x * blockDim.x + threadIdx.x;
  if (e < E) atomicAdd(&counts[src[e]], 1.f);
}

__global__ void k_coef(const float* __restrict__ L, const int* __restrict__ src,
                       const int* __restrict__ dst, int E, int D, float* __restrict__ coef)
{
  int e = blockIdx.x * blockDim.x + threadIdx.x;
  if (e >= E) return;
  const float* Li = L + (long)src[e] * D;
  const float* Lj = L + (long)dst[e] * D;
  float num = 0.f, den = 1e-10f;
  for (int d = 0; d < D; ++d) { num += Li[d] * Lj[d]; den += Lj[d] * Lj[d]; }
  coef[e] = num / den;
}

__global__ void k_agg(const float* __restrict__ L, const int* __restrict__ src,
                      const int* __restrict__ dst, const float* __restrict__ coef,
                      int E, int D, float* __restrict__ agg)
{
  long i = (long)blockIdx.x * blockDim.x + threadIdx.x;
  if (i >= (long)E * D) return;
  int e = (int)(i / D), d = (int)(i % D);
  atomicAdd(&agg[(long)src[e] * D + d], coef[e] * L[(long)dst[e] * D + d]);
}

__global__ void k_decomp_update(float* __restrict__ L, const float* __restrict__ agg,
                                const float* __restrict__ counts, int n, int D)
{
  int i = blockIdx.x * blockDim.x + threadIdx.x;
  if (i >= n * D) return;
  int node = i / D;
  float cnt = counts[node];
  if (cnt > 0.f) L[i] = L[i] - agg[i] / fmaxf(cnt, 1.f);
}

// ------------------------------ attention -----------------------------------
__global__ void k_rowmax(const float* __restrict__ S, int rows, int ldS, int n,
                         float* __restrict__ rm)
{
  int r = blockIdx.x * blockDim.x + threadIdx.x;
  if (r >= rows) return;
  float m = -INFINITY;
  for (int i = 0; i < n; ++i) m = fmaxf(m, S[(long)r * ldS + i]);
  rm[r] = m;
}

__global__ void k_softmax_T(const float* __restrict__ rm, int Bq, int T,
                            float* __restrict__ att)
{
  int b = threadIdx.x;
  if (b >= Bq) return;
  const float* x = rm + (long)b * T;
  float m = -INFINITY;
  for (int t = 0; t < T; ++t) m = fmaxf(m, x[t]);
  float s = 0.f;
  for (int t = 0; t < T; ++t) { float e = expf(x[t] - m); att[(long)b * T + t] = e; s += e; }
  float inv = 1.f / s;
  for (int t = 0; t < T; ++t) att[(long)b * T + t] *= inv;
}

__global__ void k_wsum(const float* __restrict__ att, const float* __restrict__ dh,
                       int Bq, int T, int D, float* __restrict__ out)
{
  int i = blockIdx.x * blockDim.x + threadIdx.x;
  if (i >= Bq * D) return;
  int b = i / D, d = i - b * D;
  float s = 0.f;
  for (int t = 0; t < T; ++t) s += att[(long)b * T + t] * dh[((long)b * T + t) * D + d];
  out[i] = s;
}

__global__ void k_mask_softmax_p16(const float* __restrict__ S, int rows, int ldS,
                                   int Vn, int Vp, _Float16* __restrict__ p16)
{
  int r = blockIdx.x * blockDim.x + threadIdx.x;
  if (r >= rows) return;
  const float* x = S + (long)r * ldS;
  _Float16* p = p16 + (long)r * Vp;
  float m = -INFINITY;
  for (int v = 0; v < Vn; ++v) { float a = x[v]; if (a != 0.f) m = fmaxf(m, a); }
  if (m == -INFINITY) { for (int v = 0; v < Vp; ++v) p[v] = (_Float16)0.f; return; }
  float tmp[64];
  float s = 0.f;
  for (int v = 0; v < Vn; ++v) {
    float a = x[v];
    float e = (a != 0.f) ? expf(a - m) : 0.f;
    tmp[v] = e; s += e;
  }
  float inv = 1.f / s;
  for (int v = 0; v < Vn; ++v) p[v] = (_Float16)(tmp[v] * inv);
  for (int v = Vn; v < Vp; ++v) p[v] = (_Float16)0.f;
}

__global__ void k_fact_sep(const float* __restrict__ circ, const float* __restrict__ scen,
                           int rows, int D, float* __restrict__ adc, float* __restrict__ sec)
{
  int r = blockIdx.x * blockDim.x + threadIdx.x;
  if (r >= rows) return;
  const float* c = circ + (long)r * D;
  const float* s = scen + (long)r * D;
  float x3 = 0.f, x4 = 1e-10f;
  for (int d = 0; d < D; ++d) { x3 += c[d] * s[d]; x4 += s[d] * s[d]; }
  float k = x3 / x4;
  float* a = adc + (long)r * D;
  float* e = sec + (long)r * D;
  for (int d = 0; d < D; ++d) { float v = k * s[d]; a[d] = v; e[d] = c[d] - v; }
}

__global__ void k_argmax(const float* __restrict__ x, int Bq, int N, int ld,
                         int* __restrict__ out)
{
  int b = threadIdx.x;
  if (b >= Bq) return;
  float m = -INFINITY; int mi = 0;
  for (int i = 0; i < N; ++i) { float v = x[(long)b * ld + i]; if (v > m) { m = v; mi = i; } }
  out[b] = mi;
}

// ===========================================================================
extern "C" void kernel_launch(void* const* d_in, const int* in_sizes, int n_in,
                              void* d_out, int out_size, void* d_ws, size_t ws_size,
                              hipStream_t stream)
{
  (void)in_sizes; (void)n_in; (void)out_size; (void)ws_size;

  const float* E       = (const float*)d_in[0];
  const int* charge_tokens  = (const int*)d_in[1];
  const int* article_tokens = (const int*)d_in[2];
  const int* documents      = (const int*)d_in[3];
  const int* charge_vt      = (const int*)d_in[4];
  const int* article_vt     = (const int*)d_in[5];
  const int* c_src = (const int*)d_in[6];
  const int* c_dst = (const int*)d_in[7];
  const int* a_src = (const int*)d_in[8];
  const int* a_dst = (const int*)d_in[9];
  const float* enc_Wih   = (const float*)d_in[10];
  const float* enc_Whh   = (const float*)d_in[11];
  const float* enc_bih   = (const float*)d_in[12];
  const float* enc_bhh   = (const float*)d_in[13];
  const float* encch_Wih = (const float*)d_in[14];
  const float* encch_Whh = (const float*)d_in[15];
  const float* encch_bih = (const float*)d_in[16];
  const float* encch_bhh = (const float*)d_in[17];
  const float* term_Wih  = (const float*)d_in[18];
  const float* term_Whh  = (const float*)d_in[19];
  const float* term_bih  = (const float*)d_in[20];
  const float* term_bhh  = (const float*)d_in[21];
  const float* art_Wih   = (const float*)d_in[22];
  const float* art_Whh   = (const float*)d_in[23];
  const float* art_bih   = (const float*)d_in[24];
  const float* art_bhh   = (const float*)d_in[25];
  const float* W_charge  = (const float*)d_in[26];
  const float* b_charge  = (const float*)d_in[27];
  const float* W_article = (const float*)d_in[28];
  const float* b_article = (const float*)d_in[29];
  const float* W_time    = (const float*)d_in[30];
  const float* b_time    = (const float*)d_in[31];
  float* out = (float*)d_out;

  hipFuncSetAttribute((const void*)k_gru_scan,
                      hipFuncAttributeMaxDynamicSharedMemorySize, 300 * 1024);
  hipFuncSetAttribute((const void*)k_gemm,
                      hipFuncAttributeMaxDynamicSharedMemorySize, 64 * 1024);

  // ---- workspace arena ----
  size_t off = 0; char* base = (char*)d_ws;
  auto alloc = [&](size_t bytes) -> void* {
    void* p = base + off; off = (off + bytes + 255) & ~(size_t)255; return p;
  };
  auto af16 = [&](size_t e) -> _Float16* { return (_Float16*)alloc(e * 2); };
  auto af32 = [&](size_t e) -> float*    { return (float*)alloc(e * 4); };
  auto ai32 = [&](size_t e) -> int*      { return (int*)alloc(e * 4); };

  // B operands of k_gemm: rows padded to 64.  Whh (scan-only): rows to 16.
  _Float16 *encch_Wih16[2], *encch_Whh16[2], *enc_Wih16[2], *enc_Whh16[2];
  _Float16 *term_Wih16[2], *term_Whh16[2], *art_Wih16[2], *art_Whh16[2];
  for (int d = 0; d < 2; ++d) {
    encch_Wih16[d] = af16((size_t)512 * 224);
    encch_Whh16[d] = af16((size_t)464 * 160);
    enc_Wih16[d]   = af16((size_t)512 * 224);
    enc_Whh16[d]   = af16((size_t)464 * 160);
    term_Wih16[d]  = af16((size_t)1408 * 928);
    term_Whh16[d]  = af16((size_t)1360 * 480);
    art_Wih16[d]   = af16((size_t)1856 * 1216);
    art_Whh16[d]   = af16((size_t)1808 * 608);
  }
  _Float16* Wc16 = af16((size_t)128 * 928);
  _Float16* Wa16 = af16((size_t)128 * 1216);
  _Float16* Wt16 = af16((size_t)64 * 928);

  _Float16* xch16  = af16((size_t)5952 * 224);
  _Float16* xar16  = af16((size_t)5152 * 224);
  _Float16* xdoc16 = af16((size_t)16384 * 224);
  _Float16* xv16   = af16((size_t)1600 * 224);
  float* xs_f = af32((size_t)16384 * 1800);
  float* xs_b = af32((size_t)16384 * 1800);
  float* ch_h = af32((size_t)5950 * 300);
  float* ar_h = af32((size_t)5150 * 300);
  float* d_hidden = af32((size_t)16384 * 300);
  float* vh   = af32((size_t)1600 * 300);
  float* ysbig = af32((size_t)16384 * 1200);
  float* chg = af32((size_t)CNC * 300);
  float* artm = af32((size_t)CNA * 300);
  float* new_charge = af32((size_t)CNC * 300);
  float* new_article = af32((size_t)CNA * 300);
  float* counts = af32(128);
  float* agg = af32((size_t)CNC * 300);
  float* coefb = af32(CEC);
  float* dmean = af32((size_t)32 * 300);
  float* cwc = af32((size_t)32 * 300);
  float* cwa = af32((size_t)32 * 300);
  float* cwna = af32((size_t)32 * 300);
  float* cwb = af32((size_t)32 * 300);
  float* Sbuf = af32((size_t)16384 * 128);
  float* rm  = af32(16384);
  float* att = af32((size_t)32 * 512);
  float* scen = af32((size_t)16384 * 300);
  float* adc  = af32((size_t)16384 * 300);
  float* secv = af32((size_t)16384 * 300);
  float* ssc  = af32((size_t)16384 * 300);
  float* dsc  = af32((size_t)16384 * 300);
  _Float16* dh16  = af16((size_t)16384 * 320);
  _Float16* sec16 = af16((size_t)16384 * 320);
  _Float16* q16   = af16((size_t)128 * 320);
  _Float16* vh16  = af16((size_t)32 * 64 * 320);
  _Float16* vhT16 = af16((size_t)32 * 320 * 64);
  _Float16* p16   = af16((size_t)16384 * 64);
  _Float16* df16  = af16((size_t)32 * 928);
  _Float16* fam16 = af16((size_t)32 * 1216);
  _Float16* thm16 = af16((size_t)32 * 928);
  _Float16* fa16  = af16((size_t)16384 * 1216);
  _Float16* tm16  = af16((size_t)16384 * 928);
  float* famean = af32((size_t)32 * 1200);
  float* thmean = af32((size_t)32 * 900);
  int* c_pred = ai32(32);
  int* a_pred = ai32(32);

  // ---- launch helpers ----
  auto pad16 = [&](const float* src, int R, int Cs, _Float16* dst, int Rp, int Cp) {
    long n = (long)Rp * Cp;
    k_pad16<<<(unsigned)((n + 255) / 256), 256, 0, stream>>>(src, R, Cs, dst, Rp, Cp);
  };
  auto gemm = [&](const _Float16* A, const _Float16* Bm, const float* bias, float* C,
                  int M, int N, int Kp, int lda, int ldb, int ldc,
                  long sA, long sB, long sC, int batches) {
    int groups = ceil64i(N) / 64;
    dim3 blk(32, 4);
    dim3 grd((unsigned)((groups + 3) / 4), (unsigned)(ceil16i(M) / 16), (unsigned)batches);
    size_t sh = (size_t)32 * Kp;                       // 16 rows * Kp halves
    k_gemm<<<grd, blk, sh, stream>>>(A, Bm, bias, C, M, N, Kp, lda, ldb, ldc, sA, sB, sC);
  };
  auto scan = [&](const float* xf, const float* xb, const _Float16* Wf, const _Float16* Wb,
                  const float* bhh2, float* ys, int Bseq, int T, int Hd) {
    int Kp = ceil32i(Hd), Np = ceil16i(3 * Hd), Bp = ceil16i(Bseq);
    size_t sh = (size_t)Bp * Kp * 2 + (size_t)Bp * Np * 4;
    k_gru_scan<<<2, dim3(32, 32), sh, stream>>>(xf, xb, Wf, Wb, bhh2, ys, Bseq, T, Hd, Kp, Np);
  };
  auto meanT = [&](const float* in, int Bq, int T, int ld, int C, float* o) {
    k_meanT<<<(unsigned)((Bq * C + 255) / 256), 256, 0, stream>>>(in, Bq, T, ld, C, o);
  };
  auto cat16 = [&](const float* s0, int w0, int b0, const float* s1, int w1, int b1,
                   const float* s2, int w2, int b2, const float* s3, int w3, int b3,
                   int Mrows, int T, _Float16* dst, int Cp) {
    long n = (long)ceil16i(Mrows) * Cp;
    k_cat16<<<(unsigned)((n + 255) / 256), 256, 0, stream>>>(
        s0, w0, b0, s1, w1, b1, s2, w2, b2, s3, w3, b3, Mrows, T, dst, Cp);
  };

  // ---- phase 0: stage all weights to padded f16 ----
  for (int d = 0; d < 2; ++d) {
    pad16(encch_Wih + (long)d * 450 * 200, 450, 200, encch_Wih16[d], 512, 224);
    pad16(encch_Whh + (long)d * 450 * 150, 450, 150, encch_Whh16[d], 464, 160);
    pad16(enc_Wih   + (long)d * 450 * 200, 450, 200, enc_Wih16[d],   512, 224);
    pad16(enc_Whh   + (long)d * 450 * 150, 450, 150, enc_Whh16[d],   464, 160);
    pad16(term_Wih  + (long)d * 1350 * 900, 1350, 900, term_Wih16[d], 1408, 928);
    pad16(term_Whh  + (long)d * 1350 * 450, 1350, 450, term_Whh16[d], 1360, 480);
    pad16(art_Wih   + (long)d * 1800 * 1200, 1800, 1200, art_Wih16[d], 1856, 1216);
    pad16(art_Whh   + (long)d * 1800 * 600, 1800, 600, art_Whh16[d],  1808, 608);
  }
  pad16(W_charge, CNC, 900, Wc16, 128, 928);
  pad16(W_article, CNA, 1200, Wa16, 128, 1216);
  pad16(W_time, CNT, 900, Wt16, 64, 928);

  // ---- phase 1/2: charge & article token encoders (encch) ----
  {
    long n = (long)5952 * 224;
    k_embed16<<<(unsigned)((n + 255) / 256), 256, 0, stream>>>(E, charge_tokens, CNC * CLC, xch16, 5952, 224);
  }
  gemm(xch16, encch_Wih16[0], encch_bih,       xs_f, 5950, 450, 224, 224, 224, 450, 0, 0, 0, 1);
  gemm(xch16, encch_Wih16[1], encch_bih + 450, xs_b, 5950, 450, 224, 224, 224, 450, 0, 0, 0, 1);
  scan(xs_f, xs_b, encch_Whh16[0], encch_Whh16[1], encch_bhh, ch_h, CNC, CLC, CHID);
  meanT(ch_h, CNC, CLC, 300, 300, chg);

  {
    long n = (long)5152 * 224;
    k_embed16<<<(unsigned)((n + 255) / 256), 256, 0, stream>>>(E, article_tokens, CNA * CLC, xar16, 5152, 224);
  }
  gemm(xar16, encch_Wih16[0], encch_bih,       xs_f, 5150, 450, 224, 224, 224, 450, 0, 0, 0, 1);
  gemm(xar16, encch_Wih16[1], encch_bih + 450, xs_b, 5150, 450, 224, 224, 224, 450, 0, 0, 0, 1);
  scan(xs_f, xs_b, encch_Whh16[0], encch_Whh16[1], encch_bhh, ar_h, CNA, CLC, CHID);
  meanT(ar_h, CNA, CLC, 300, 300, artm);

  // ---- phase 3: graph decomposition ----
  auto decomp = [&](float* L, const int* src, const int* dst, int n, int Earr) {
    hipMemsetAsync(counts, 0, 128 * 4, stream);
    k_counts<<<(unsigned)((Earr + 255) / 256), 256, 0, stream>>>(src, Earr, counts);
    for (int layer = 0; layer < 2; ++layer) {
      k_coef<<<(unsigned)((Earr + 255) / 256), 256, 0, stream>>>(L, src, dst, Earr, 300, coefb);
      hipMemsetAsync(agg, 0, (size_t)n * 300 * 4, stream);
      long na = (long)Earr * 300;
      k_agg<<<(unsigned)((na + 255) / 256), 256, 0, stream>>>(L, src, dst, coefb, Earr, 300, agg);
      k_decomp_update<<<(unsigned)((n * 300 + 255) / 256), 256, 0, stream>>>(L, agg, counts, n, 300);
    }
  };
  hipMemcpyAsync(new_charge, chg, (size_t)CNC * 300 * 4, hipMemcpyDeviceToDevice, stream);
  hipMemcpyAsync(new_article, artm, (size_t)CNA * 300 * 4, hipMemcpyDeviceToDevice, stream);
  decomp(new_charge, c_src, c_dst, CNC, CEC);
  decomp(new_article, a_src, a_dst, CNA, CEA);

  // ---- phase 4: document encoder ----
  {
    long n = (long)16384 * 224;
    k_embed16<<<(unsigned)((n + 255) / 256), 256, 0, stream>>>(E, documents, CB * CT, xdoc16, 16384, 224);
  }
  gemm(xdoc16, enc_Wih16[0], enc_bih,       xs_f, 16384, 450, 224, 224, 224, 450, 0, 0, 0, 1);
  gemm(xdoc16, enc_Wih16[1], enc_bih + 450, xs_b, 16384, 450, 224, 224, 224, 450, 0, 0, 0, 1);
  scan(xs_f, xs_b, enc_Whh16[0], enc_Whh16[1], enc_bhh, d_hidden, CB, CT, CHID);
  pad16(d_hidden, 16384, 300, dh16, 16384, 320);
  meanT(d_hidden, CB, CT, 300, 300, dmean);

  // ---- code_wise attention helper ----
  auto code_wise = [&](const float* q, int n, float* o) {
    pad16(q, n, 300, q16, 128, 320);
    gemm(dh16, q16, nullptr, Sbuf, 16384, n, 320, 320, 320, n, 0, 0, 0, 1);
    k_rowmax<<<(unsigned)((16384 + 255) / 256), 256, 0, stream>>>(Sbuf, 16384, n, n, rm);
    k_softmax_T<<<1, 32, 0, stream>>>(rm, CB, CT, att);
    k_wsum<<<(unsigned)((CB * 300 + 255) / 256), 256, 0, stream>>>(att, d_hidden, CB, CT, 300, o);
  };
  code_wise(new_charge, CNC, cwc);
  code_wise(chg, CNC, cwa);

  // ---- phase 6: charge classifier ----
  cat16(dmean, 300, 0, cwc, 300, 0, cwa, 300, 0, nullptr, 0, 0, 32, 1, df16, 928);
  gemm(df16, Wc16, b_charge, out, 32, CNC, 928, 928, 928, CNC, 0, 0, 0, 1);
  k_argmax<<<1, 32, 0, stream>>>(out, 32, CNC, CNC, c_pred);

  // ---- fact separation helper ----
  auto fact_sep = [&](const int* vt, const int* pred, const float* circ,
                      const _Float16* circ16, float* adc_o, float* sec_o) {
    {
      long n = (long)1600 * 224;
      k_embed_pred16<<<(unsigned)((n + 255) / 256), 256, 0, stream>>>(E, vt, pred, CLV, xv16, 1600, 224);
    }
    gemm(xv16, enc_Wih16[0], enc_bih,       xs_f, 1600, 450, 224, 224, 224, 450, 0, 0, 0, 1);
    gemm(xv16, enc_Wih16[1], enc_bih + 450, xs_b, 1600, 450, 224, 224, 224, 450, 0, 0, 0, 1);
    scan(xs_f, xs_b, enc_Whh16[0], enc_Whh16[1], enc_bhh, vh, CB, CLV, CHID);
    {
      long n = (long)32 * 64 * 320;
      k_pad_batch16<<<(unsigned)((n + 255) / 256), 256, 0, stream>>>(vh, 32, CLV, 300, vh16, 64, 320, 0);
    }
    gemm(circ16, vh16, nullptr, Sbuf, 512, CLV, 320, 320, 320, 64,
         (long)512 * 320, (long)64 * 320, (long)512 * 64, 32);
    k_mask_softmax_p16<<<(unsigned)((16384 + 255) / 256), 256, 0, stream>>>(Sbuf, 16384, 64, CLV, 64, p16);
    {
      long n = (long)32 * 320 * 64;
      k_pad_batch16<<<(unsigned)((n + 255) / 256), 256, 0, stream>>>(vh, 32, CLV, 300, vhT16, 320, 64, 1);
    }
    gemm(p16, vhT16, nullptr, scen, 512, 300, 64, 64, 64, 300,
         (long)512 * 64, (long)320 * 64, (long)512 * 300, 32);
    k_fact_sep<<<(unsigned)((16384 + 255) / 256), 256, 0, stream>>>(circ, scen, 16384, 300, adc_o, sec_o);
  };
  fact_sep(charge_vt, c_pred, d_hidden, dh16, adc, secv);

  // ---- phase 8: article-side code_wise ----
  code_wise(new_article, CNA, cwna);
  code_wise(artm, CNA, cwb);

  // ---- phase 9: article BiGRU + classifier ----
  cat16(d_hidden, 300, 0, cwna, 300, 1, adc, 300, 0, cwb, 300, 1, 16384, CT, fa16, 1216);
  gemm(fa16, art_Wih16[0], art_bih,        xs_f, 16384, 1800, 1216, 1216, 1216, 1800, 0, 0, 0, 1);
  gemm(fa16, art_Wih16[1], art_bih + 1800, xs_b, 16384, 1800, 1216, 1216, 1216, 1800, 0, 0, 0, 1);
  scan(xs_f, xs_b, art_Whh16[0], art_Whh16[1], art_bhh, ysbig, CB, CT, 600);
  meanT(ysbig, CB, CT, 1200, 1200, famean);
  pad16(famean, 32, 1200, fam16, 32, 1216);
  gemm(fam16, Wa16, b_article, out + 32 * CNC, 32, CNA, 1216, 1216, 1216, CNA, 0, 0, 0, 1);
  k_argmax<<<1, 32, 0, stream>>>(out + 32 * CNC, 32, CNA, CNA, a_pred);

  // ---- phase 10: second fact separation (context = sec_vector) ----
  pad16(secv, 16384, 300, sec16, 16384, 320);
  fact_sep(article_vt, a_pred, secv, sec16, ssc, dsc);

  // ---- phase 11: term BiGRU + time classifier ----
  cat16(d_hidden, 300, 0, ssc, 300, 0, dsc, 300, 0, nullptr, 0, 0, 16384, CT, tm16, 928);
  gemm(tm16, term_Wih16[0], term_bih,        xs_f, 16384, 1350, 928, 928, 928, 1350, 0, 0, 0, 1);
  gemm(tm16, term_Wih16[1], term_bih + 1350, xs_b, 16384, 1350, 928, 928, 928, 1350, 0, 0, 0, 1);
  scan(xs_f, xs_b, term_Whh16[0], term_Whh16[1], term_bhh, ysbig, CB, CT, 450);
  meanT(ysbig, CB, CT, 900, 900, thmean);
  pad16(thmean, 32, 900, thm16, 32, 928);
  gemm(thm16, Wt16, b_time, out + 32 * CNC + 32 * CNA, 32, CNT, 928, 928, 928, CNT, 0, 0, 0, 1);
}
